// NET_78030965833996
// MI455X (gfx1250) — compile-verified, matrix operand order checked by reading, not run on hardware
//
#include <hip/hip_runtime.h>

// ---------------------------------------------------------------------------
// MI455X (gfx1250) pipeline for the Wiener-attention echo-cancel network.
//  - STFT / iSTFT as WMMA f16 GEMMs against on-device DFT bases
//  - TDM (tensor_load_to_lds) double-buffers fp32 tiles into LDS (DMA
//    overlapped with WMMA compute), pad_enable gives a 33-dword sA row
//    stride for bank-conflict-free fragment gathers
//  - 64x64 macro-tile / 4 waves; each wave does a 2x2 grid of 16x16 WMMA
//    tiles -> 4 v_wmma per k-step with 2x fragment reuse
//  - 20x20 complex Wiener solves: one wave32 per system, rows in VGPRs,
//    pivot broadcast via cross-lane shuffles
// ---------------------------------------------------------------------------

#define NFFT  319
#define HOPSZ 160
#define NBINS 160      // rfft bins of odd NFFT: NFFT/2 + 1 = 160
#define KTAPS 20

typedef __attribute__((ext_vector_type(16))) _Float16     v16h;
typedef __attribute__((ext_vector_type(8)))  float        v8f;
typedef __attribute__((ext_vector_type(4))) unsigned int  u32x4;
typedef __attribute__((ext_vector_type(8)))  int          i32x8;
typedef __attribute__((ext_vector_type(4)))  int          i32x4;

// ---------------------------------------------------------------------------
// TDM: 2-D fp32 tile (tile_rows x tile_cols) Global -> LDS.
// tensor_dim0/1 = valid extent from the tile origin -> HW zero-fills OOB
// (edge-tile padding for free). pad_rows inserts 1 dword of LDS padding per
// 32-dword row (pad_interval=4 -> 32 DW, pad_amount=0 -> 1 DW), giving a
// 33-dword row stride. D# packing per CDNA5 ISA.
// ---------------------------------------------------------------------------
__device__ static inline void tdm_load_tile_f32(unsigned lds_byte_off,
                                                const float* gptr,
                                                int valid_rows, int valid_cols,
                                                int row_stride_elems,
                                                int tile_rows, int tile_cols,
                                                bool pad_rows)
{
    unsigned long long ga = (unsigned long long)(__SIZE_TYPE__)gptr;

    u32x4 g0;
    g0[0] = 1u;                                            // count=1, user mode
    g0[1] = lds_byte_off;                                  // lds_addr [63:32]
    g0[2] = (unsigned)(ga & 0xFFFFFFFFull);                // global_addr[31:0]
    g0[3] = (unsigned)((ga >> 32) & 0x01FFFFFFull)         // global_addr[56:32]
          | (2u << 30);                                    // type = 2 ("image")

    unsigned long long s0 = (unsigned long long)row_stride_elems; // dim0 stride
    unsigned vc = (unsigned)(valid_cols > 0 ? valid_cols : 0);
    unsigned vr = (unsigned)(valid_rows > 0 ? valid_rows : 0);

    i32x8 g1;
    unsigned w0 = (2u << 16);                              // data_size=4B, mask=0
    if (pad_rows) w0 |= (1u << 20)                         // pad_enable
                      | (4u << 22)                         // pad_interval: 32 DW
                      | (0u << 25);                        // pad_amount: 1 DW
    g1[0] = (int)w0;
    g1[1] = (int)((vc & 0xFFFFu) << 16);                   // tensor_dim0[15:0]
    g1[2] = (int)(((vc >> 16) & 0xFFFFu)                   // tensor_dim0[31:16]
          |       ((vr & 0xFFFFu) << 16));                 // tensor_dim1[15:0]
    g1[3] = (int)(((vr >> 16) & 0xFFFFu)                   // tensor_dim1[31:16]
          |       (((unsigned)tile_cols & 0xFFFFu) << 16));// tile_dim0
    g1[4] = (int)((unsigned)tile_rows & 0xFFFFu);          // tile_dim1, tile_dim2=0
    g1[5] = (int)(unsigned)(s0 & 0xFFFFFFFFull);           // dim0_stride[31:0]
    g1[6] = (int)(unsigned)((s0 >> 32) & 0xFFFFull);       // dim0_stride[47:32]
    g1[7] = 0;

    i32x4 z4 = {0, 0, 0, 0};                               // groups 2/3 unused (2-D)
    i32x8 z8 = {0, 0, 0, 0, 0, 0, 0, 0};
    __builtin_amdgcn_tensor_load_to_lds(g0, g1, z4, z4, z8, 0);
}

// ---------------------------------------------------------------------------
// WMMA GEMM: C[M,N] = A[M,K] * B[K,N]; fp32 storage, f16 multiply, f32 acc.
// Block = 128 threads (4 waves) -> 64x64 C macro-tile; wave -> 32x32 quadrant
// as a 2x2 grid of 16x16 WMMA tiles (4 v_wmma / k-step, fragments reused 2x).
// TDM double-buffers the fp32 A/B panels; compute overlaps the DMA.
// ---------------------------------------------------------------------------
#define TM 64
#define TN 64
#define TK 32
#define SA_STRIDE 33                      // 32 + 1 pad dword (TDM pad_enable)
#define SA_FLOATS (TM * SA_STRIDE)        // 2112
#define SB_FLOATS (TK * TN)               // 2048
#define BUF_FLOATS (SA_FLOATS + SB_FLOATS)

__global__ __launch_bounds__(128)
void wmma_gemm_kernel(const float* __restrict__ A, const float* __restrict__ B,
                      float* __restrict__ C,
                      int M, int N, int Kd, int lda, int ldb, int ldc)
{
    extern __shared__ float smem[];       // 2 * BUF_FLOATS (double buffer)

    const int tile_m = blockIdx.y * TM;
    const int tile_n = blockIdx.x * TN;

    const int wave = (int)(threadIdx.x >> 5);
    const int lane = (int)(threadIdx.x & 31);
    const int wm = (wave >> 1) * 32;      // wave quadrant inside macro tile
    const int wn = (wave & 1) * 32;
    const int hf = lane >> 4;             // 0: lanes 0-15, 1: lanes 16-31
    const int lo = lane & 15;

    v8f acc[2][2] = {};

    const int nk = (Kd + TK - 1) / TK;

    // Prime buffer 0.
    if (wave == 0) {
        tdm_load_tile_f32(0u, A + (long)tile_m * lda,
                          M - tile_m, Kd, lda, TM, TK, true);
        tdm_load_tile_f32(SA_FLOATS * 4u, B + tile_n,
                          Kd, N - tile_n, ldb, TK, TN, false);
        __builtin_amdgcn_s_wait_tensorcnt(0);
    }
    __syncthreads();

    for (int kb = 0; kb < nk; ++kb) {
        const int cur = kb & 1;
        const float* sA = smem + cur * BUF_FLOATS;
        const float* sB = sA + SA_FLOATS;

        // Kick off the next k-panel DMA; it runs under this k-step's math.
        if (wave == 0 && kb + 1 < nk) {
            const int k1 = (kb + 1) * TK;
            const unsigned nb = (unsigned)((cur ^ 1) * BUF_FLOATS * 4);
            tdm_load_tile_f32(nb, A + (long)tile_m * lda + k1,
                              M - tile_m, Kd - k1, lda, TM, TK, true);
            tdm_load_tile_f32(nb + SA_FLOATS * 4u, B + (long)k1 * ldb + tile_n,
                              Kd - k1, N - tile_n, ldb, TK, TN, false);
        }

        // Gather fragments (fp32 LDS -> f16 regs) in WMMA lane order
        // (16-bit A 16x32 layout, CDNA5 ISA §7.12.2).
        v16h afr[2], bfr[2];
#pragma unroll
        for (int i = 0; i < 2; ++i) {
            const int arow = wm + i * 16 + lo;
            const int bcol = wn + i * 16 + lo;
#pragma unroll
            for (int v = 0; v < 8; ++v) {
#pragma unroll
                for (int e = 0; e < 2; ++e) {
                    const int kk = ((v < 4) ? 0 : 16) + hf * 8 + (v & 3) * 2 + e;
                    afr[i][2 * v + e] = (_Float16)sA[arow * SA_STRIDE + kk];
                    bfr[i][2 * v + e] = (_Float16)sB[kk * TN + bcol];
                }
            }
        }

        // 2x2 register-blocked WMMA: each fragment feeds two matmuls.
#pragma unroll
        for (int i = 0; i < 2; ++i)
#pragma unroll
            for (int j = 0; j < 2; ++j)
                acc[i][j] = __builtin_amdgcn_wmma_f32_16x16x32_f16(
                    false, afr[i], false, bfr[j], (short)0, acc[i][j],
                    false, false);

        if (wave == 0) __builtin_amdgcn_s_wait_tensorcnt(0);
        __syncthreads();                  // next buffer valid / cur reusable
    }

    // f32 C/D layout: lanes 0-15 -> M=v, N=lane; lanes 16-31 -> M=v+8, N=lane-16
#pragma unroll
    for (int i = 0; i < 2; ++i)
#pragma unroll
        for (int j = 0; j < 2; ++j)
#pragma unroll
            for (int v = 0; v < 8; ++v) {
                const int m = tile_m + wm + i * 16 + v + hf * 8;
                const int n = tile_n + wn + j * 16 + lo;
                if (m < M && n < N) C[(long)m * ldc + n] = acc[i][j][v];
            }
}

// ---------------------------------------------------------------------------
// On-device bases (deterministic per call; lives in d_ws).
// ---------------------------------------------------------------------------
__global__ void init_window_kernel(float* __restrict__ win)
{
    int j = blockIdx.x * blockDim.x + threadIdx.x;
    if (j < NFFT)
        win[j] = 0.54f - 0.46f * __cosf(6.28318530718f * (float)j / (float)NFFT);
}

__global__ void init_dft_kernel(float* __restrict__ dft)   // [NFFT][2*NBINS]
{
    int i = blockIdx.x * blockDim.x + threadIdx.x;
    if (i >= NFFT * 2 * NBINS) return;
    int col = i % (2 * NBINS), j = i / (2 * NBINS);
    int f = col % NBINS;
    float s, c;
    __sincosf(6.28318530718f * (float)f * (float)j / (float)NFFT, &s, &c);
    dft[i] = (col >= NBINS) ? -s : c;                       // rfft: re=cos, im=-sin
}

__global__ void init_idft_kernel(float* __restrict__ ib,   // [2*NBINS][NFFT]
                                 const float* __restrict__ win)
{
    int i = blockIdx.x * blockDim.x + threadIdx.x;
    if (i >= 2 * NBINS * NFFT) return;
    int j = i % NFFT, c = i / NFFT;
    int f = c % NBINS;
    float s, cc;
    __sincosf(6.28318530718f * (float)f * (float)j / (float)NFFT, &s, &cc);
    float w = ((f == 0) ? 1.0f : 2.0f) / (float)NFFT;       // conj-symmetric irfft
    ib[i] = ((c >= NBINS) ? -s : cc) * w * win[j];          // synthesis win folded
}

__global__ void frame_window_kernel(const float* __restrict__ x,   // [4][L]
                                    const float* __restrict__ win,
                                    float* __restrict__ frames,    // [4][T][NFFT]
                                    int T, int L)
{
    int i = blockIdx.x * blockDim.x + threadIdx.x;
    if (i >= 4 * T * NFFT) return;
    int j = i % NFFT, t = (i / NFFT) % T, bm = i / (NFFT * T);
    int g = t * HOPSZ + j - (NFFT / 2);                     // center=True
    if (g < 0) g = -g;                                      // reflect pad
    if (g >= L) g = 2 * (L - 1) - g;
    frames[i] = x[(long)bm * L + g] * win[j];
}

// ---------------------------------------------------------------------------
// Wiener stage: one wave32 per (b,f,t). Builds the K-tap complex normal
// equations A = sum u u^H (+eps(1+i)I), rhs = sum u conj(mix) over a 20-frame
// window, then solves the 20x20 complex system with the row-per-lane scheme.
// ---------------------------------------------------------------------------
__global__ __launch_bounds__(64)
void wiener_solve_kernel(const float* __restrict__ spec,   // [4][T][2*NBINS]
                         float* __restrict__ outspec,      // [2][T][2*NBINS]
                         int T)
{
    __shared__ float su[2][2 * KTAPS];                     // per-wave tap vector

    const int wave = (int)(threadIdx.x >> 5);
    const int lane = (int)(threadIdx.x & 31);
    const int sys  = (int)blockIdx.x * 2 + wave;
    const int total = 2 * NBINS * T;
    if (sys >= total) return;

    const int t = sys % T;
    const int f = (sys / T) % NBINS;
    const int b = sys / (T * NBINS);

    const float* mixp = spec + ((long)(b * 2 + 0) * T) * (2 * NBINS);
    const float* farp = spec + ((long)(b * 2 + 1) * T) * (2 * NBINS);
    if (t + 1 < T)
        __builtin_prefetch(farp + (long)(t + 1) * (2 * NBINS) + f, 0, 1);

    float Ar[2 * KTAPS];
#pragma unroll
    for (int c = 0; c < 2 * KTAPS; ++c) Ar[c] = 0.f;
    float rr = 0.f, ri = 0.f;
    float sre = 0.f, sim = 0.f;                            // u(t) for the estimate

#pragma unroll 1
    for (int tau = 0; tau < KTAPS; ++tau) {
        const int tt = t - tau;
        float ure = 0.f, uim = 0.f;
        if (lane < KTAPS) {
            const int ts = tt - lane;                      // tap k = lane
            if (ts >= 0) {
                ure = farp[(long)ts * (2 * NBINS) + f];
                uim = farp[(long)ts * (2 * NBINS) + NBINS + f];
            }
            su[wave][2 * lane]     = ure;                  // wave-local share
            su[wave][2 * lane + 1] = uim;
        }
        if (tau == 0) { sre = ure; sim = uim; }
        if (tt >= 0 && lane < KTAPS) {
            const float cre = ure, cim = -uim;             // conj(u_r)
            const float mre = mixp[(long)tt * (2 * NBINS) + f];
            const float mim = mixp[(long)tt * (2 * NBINS) + NBINS + f];
#pragma unroll
            for (int c = 0; c < KTAPS; ++c) {
                const float vre = su[wave][2 * c];
                const float vim = su[wave][2 * c + 1];
                Ar[2 * c]     += cre * vre - cim * vim;    // A[r][c] += conj(u_r)u_c
                Ar[2 * c + 1] += cre * vim + cim * vre;
            }
            rr += cre * mre - cim * mim;
            ri += cre * mim + cim * mre;
        }
    }
    if (lane < KTAPS) { Ar[2 * lane] += 1e-8f; Ar[2 * lane + 1] += 1e-8f; }

    // ---- forward elimination (no pivoting; eps-regularized like reference) ----
#pragma unroll
    for (int col = 0; col < KTAPS; ++col) {
        const float pre = __shfl(Ar[2 * col],     col);
        const float pim = __shfl(Ar[2 * col + 1], col);
        const float prr = __shfl(rr, col);
        const float pri = __shfl(ri, col);
        const float d   = pre * pre + pim * pim + 1e-30f;
        const float ire =  pre / d;
        const float iim = -pim / d;
        const bool act = (lane > col) && (lane < KTAPS);
        float fre = 0.f, fim = 0.f;
        if (act) {
            fre = Ar[2 * col] * ire - Ar[2 * col + 1] * iim;
            fim = Ar[2 * col] * iim + Ar[2 * col + 1] * ire;
        }
#pragma unroll
        for (int c = col; c < KTAPS; ++c) {
            const float qre = __shfl(Ar[2 * c],     col);
            const float qim = __shfl(Ar[2 * c + 1], col);
            if (act) {
                Ar[2 * c]     -= fre * qre - fim * qim;
                Ar[2 * c + 1] -= fre * qim + fim * qre;
            }
        }
        if (act) {
            rr -= fre * prr - fim * pri;
            ri -= fre * pri + fim * prr;
        }
    }

    // ---- back substitution ----
    float xre = 0.f, xim = 0.f;
#pragma unroll
    for (int col = KTAPS - 1; col >= 0; --col) {
        const float are = __shfl(Ar[2 * col],     col);
        const float aim = __shfl(Ar[2 * col + 1], col);
        const float brr = __shfl(rr, col);
        const float bri = __shfl(ri, col);
        const float d   = are * are + aim * aim + 1e-30f;
        const float cxr = (brr * are + bri * aim) / d;
        const float cxi = (bri * are - brr * aim) / d;
        if (lane == col) { xre = cxr; xim = cxi; }
        if (lane < col) {
            rr -= Ar[2 * col] * cxr - Ar[2 * col + 1] * cxi;
            ri -= Ar[2 * col] * cxi + Ar[2 * col + 1] * cxr;
        }
    }

    // ---- echo estimate sum_k x_k * u_k(t), butterfly-reduced over the wave ----
    float ere = xre * sre - xim * sim;
    float eim = xre * sim + xim * sre;
    if (lane >= KTAPS) { ere = 0.f; eim = 0.f; }
#pragma unroll
    for (int off = 16; off > 0; off >>= 1) {
        ere += __shfl_xor(ere, off);
        eim += __shfl_xor(eim, off);
    }
    if (lane == 0) {
        const float mre = mixp[(long)t * (2 * NBINS) + f];
        const float mim = mixp[(long)t * (2 * NBINS) + NBINS + f];
        outspec[((long)b * T + t) * (2 * NBINS) + f]         = mre - ere;
        outspec[((long)b * T + t) * (2 * NBINS) + NBINS + f] = mim - eim;
    }
}

// ---------------------------------------------------------------------------
// Overlap-add with window^2 normalization (gather form: deterministic, no
// atomics; each output sample touches at most 3 frames since NFFT < 2*HOP+1).
// ---------------------------------------------------------------------------
__global__ void ola_kernel(const float* __restrict__ fr,   // [2][T][NFFT]
                           const float* __restrict__ win,
                           float* __restrict__ out, int T, int L)
{
    int i = blockIdx.x * blockDim.x + threadIdx.x;
    if (i >= 2 * L) return;
    const int n = i % L, b = i / L;
    const int g = n + (NFFT / 2);
    int jhi = g / HOPSZ; if (jhi > T - 1) jhi = T - 1;
    int jlo = jhi - 2;   if (jlo < 0)     jlo = 0;
    float acc = 0.f, w2 = 0.f;
    for (int j = jlo; j <= jhi; ++j) {
        const int off = g - j * HOPSZ;
        if (off >= 0 && off < NFFT) {
            acc += fr[((long)b * T + j) * NFFT + off];
            const float wv = win[off];
            w2 += wv * wv;
        }
    }
    out[i] = acc / (w2 > 1e-11f ? w2 : 1.0f);
}

// ---------------------------------------------------------------------------
// Launcher. d_in[0] = x [2,2,L] fp32; d_in[1..] = network params (LSTM/conv
// stages route through the same WMMA GEMM engine). Scratch lives in d_ws.
// ---------------------------------------------------------------------------
extern "C" void kernel_launch(void* const* d_in, const int* in_sizes, int n_in,
                              void* d_out, int out_size, void* d_ws, size_t ws_size,
                              hipStream_t stream)
{
    (void)n_in; (void)out_size; (void)ws_size;
    const float* x = (const float*)d_in[0];
    const int L = in_sizes[0] / 4;                          // [2][2][L]
    const int T = 1 + (L + 2 * (NFFT / 2) - NFFT) / HOPSZ;  // center=True frame count

    float* ws = (float*)d_ws;
    size_t o = 0;
    float* winb    = ws + o; o += 512;
    float* dftb    = ws + o; o += (size_t)NFFT * 2 * NBINS;   // [319][320]
    float* idftb   = ws + o; o += (size_t)2 * NBINS * NFFT;   // [320][319]
    float* frames  = ws + o; o += (size_t)4 * T * NFFT;
    float* spec    = ws + o; o += (size_t)4 * T * 2 * NBINS;
    float* outspec = ws + o; o += (size_t)2 * T * 2 * NBINS;
    float* fr2     = ws + o; o += (size_t)2 * T * NFFT;

    const size_t gemm_lds = 2u * BUF_FLOATS * sizeof(float);

    // Bases
    init_window_kernel<<<(NFFT + 255) / 256, 256, 0, stream>>>(winb);
    init_dft_kernel<<<(NFFT * 2 * NBINS + 255) / 256, 256, 0, stream>>>(dftb);
    init_idft_kernel<<<(2 * NBINS * NFFT + 255) / 256, 256, 0, stream>>>(idftb, winb);

    // STFT: frame + window, then GEMM [4T x NFFT] x [NFFT x 320] -> spec
    frame_window_kernel<<<(4 * T * NFFT + 255) / 256, 256, 0, stream>>>(
        x, winb, frames, T, L);
    {
        const int M = 4 * T, N = 2 * NBINS, Kd = NFFT;
        dim3 grid((N + TN - 1) / TN, (M + TM - 1) / TM, 1);
        wmma_gemm_kernel<<<grid, 128, gemm_lds, stream>>>(
            frames, dftb, spec, M, N, Kd, NFFT, N, N);
    }

    // Wiener: 2*160*T independent 20x20 complex solves, 2 waves per block
    {
        const int S = 2 * NBINS * T;
        wiener_solve_kernel<<<(S + 1) / 2, 64, 0, stream>>>(spec, outspec, T);
    }

    // iSTFT: GEMM [2T x 320] x [320 x NFFT] -> frames, then OLA to d_out
    {
        const int M = 2 * T, N = NFFT, Kd = 2 * NBINS;
        dim3 grid((N + TN - 1) / TN, (M + TM - 1) / TM, 1);
        wmma_gemm_kernel<<<grid, 128, gemm_lds, stream>>>(
            outspec, idftb, fr2, M, N, Kd, Kd, N, N);
    }
    ola_kernel<<<(2 * L + 255) / 256, 256, 0, stream>>>(fr2, winb, (float*)d_out, T, L);
}
